// IAFF_31988916420941
// MI455X (gfx1250) — matrix operand order repaired and checked by reading, not compiled, not run on hardware
//
#include <hip/hip_runtime.h>
#include <math.h>

typedef __attribute__((ext_vector_type(16))) _Float16 v16h;
typedef __attribute__((ext_vector_type(8)))  _Float16 v8h;
typedef __attribute__((ext_vector_type(8)))  float    v8f;

constexpr int   Npts   = 6144;
constexpr int   Knbr   = 32;
constexpr int   Cch    = 32;
constexpr int   HEADS  = 4;
constexpr int   HD     = 8;
constexpr float RADIUS = 0.1125f;   // EXT/2, EXT = 1.5*6*0.025

#define DEVFN static __device__ __forceinline__

// ---------------------------------------------------------------------------
// WMMA fragment helpers (layouts per CDNA5 ISA 7.12.2, wave32)
// ---------------------------------------------------------------------------
DEVFN v8f wmma_f16(v16h a, v16h b, v8f c) {
  return __builtin_amdgcn_wmma_f32_16x16x32_f16(false, a, false, b,
                                                (short)0, c, false, false);
}

// A fragment: 16x32 f16, row-major source, lda in halves.
// lanes 0-15: row m, K {0..7 | 16..23}; lanes 16-31: row m, K {8..15 | 24..31}
DEVFN v16h load_a_frag(const _Float16* A, int lda, int row0, int k0) {
  int lane = threadIdx.x & 31;
  int m = lane & 15;
  const _Float16* p = A + (size_t)(row0 + m) * lda + k0 + ((lane & 16) ? 8 : 0);
  v8h lo = *(const v8h*)p;
  v8h hi = *(const v8h*)(p + 16);
  v16h r;
#pragma unroll
  for (int i = 0; i < 8; ++i) { r[i] = lo[i]; r[i + 8] = hi[i]; }
  return r;
}

// B fragment: 32x16 f16. Source stored transposed: Bt[col][k], ldb in halves.
// lane = column n; lanes 0-15 hold K {0..7 | 16..23}, lanes 16-31 hold K {8..15 | 24..31}.
DEVFN v16h load_b_frag(const _Float16* Bt, int ldb, int col0, int k0) {
  int lane = threadIdx.x & 31;
  int n = lane & 15;
  const _Float16* p = Bt + (size_t)(col0 + n) * ldb + k0 + ((lane & 16) ? 8 : 0);
  v8h lo = *(const v8h*)p;
  v8h hi = *(const v8h*)(p + 16);
  v16h r;
#pragma unroll
  for (int i = 0; i < 8; ++i) { r[i] = lo[i]; r[i + 8] = hi[i]; }
  return r;
}

// ---------------------------------------------------------------------------
// Wave-per-row-stripe GEMM: C[M, NT*16] = A[M,Kd](f16) * Bt[NT*16,Kd]^T + bias
// One wave owns a 16-row stripe and all NT column tiles, so each A fragment
// is loaded once and reused by NT WMMAs (arithmetic intensity * NT).
// ---------------------------------------------------------------------------
template <int NT>
__global__ void __launch_bounds__(32)
gemm16_kernel(const _Float16* __restrict__ A, const _Float16* __restrict__ Bt,
              const float* __restrict__ bias, float* __restrict__ Cf,
              _Float16* __restrict__ Ch, int Kd, int relu) {
  const int Nc = NT * 16;
  int row0 = blockIdx.x * 16;
  v8f acc[NT];
#pragma unroll
  for (int t = 0; t < NT; ++t) acc[t] = v8f{};
  for (int k0 = 0; k0 < Kd; k0 += 32) {
    v16h a = load_a_frag(A, Kd, row0, k0);
    if (k0 + 32 < Kd) {  // hide next-K latency: global_prefetch_b8
      int lane = threadIdx.x & 31;
      __builtin_prefetch(A + (size_t)(row0 + (lane & 15)) * Kd + k0 + 32, 0, 3);
      __builtin_prefetch(Bt + (size_t)(lane & 15) * Kd + k0 + 32, 0, 3);
    }
#pragma unroll
    for (int t = 0; t < NT; ++t) {
      v16h b = load_b_frag(Bt, Kd, t * 16, k0);
      acc[t] = wmma_f16(a, b, acc[t]);
    }
  }
  int lane = threadIdx.x & 31;
  int n = lane & 15;
  int rbase = row0 + ((lane & 16) ? 8 : 0);
#pragma unroll
  for (int t = 0; t < NT; ++t) {
    float bv = bias ? bias[t * 16 + n] : 0.f;
#pragma unroll
    for (int r = 0; r < 8; ++r) {
      float v = acc[t][r] + bv;
      if (relu) v = fmaxf(v, 0.f);
      size_t off = (size_t)(rbase + r) * Nc + t * 16 + n;
      if (Cf) Cf[off] = v;
      if (Ch) Ch[off] = (_Float16)v;
    }
  }
}

// ---------------------------------------------------------------------------
// Geometry: per (point, neighbor) trilinear tap weights + bin base index
// ---------------------------------------------------------------------------
__global__ void geom_kernel(const float* __restrict__ pos, const int* __restrict__ nbr,
                            float* gw, float* gf0, float* gf1, float* gf2, int* gbase) {
  int i = blockIdx.x * blockDim.x + threadIdx.x;
  if (i >= Npts * Knbr) return;
  int nn = i / Knbr;
  int j = nbr[i];
  float u0 = (pos[j * 3 + 0] - pos[nn * 3 + 0]) / RADIUS;
  float u1 = (pos[j * 3 + 1] - pos[nn * 3 + 1]) / RADIUS;
  float u2 = (pos[j * 3 + 2] - pos[nn * 3 + 2]) / RADIUS;
  float r2 = u0 * u0 + u1 * u1 + u2 * u2;
  float w = 1.f - r2; w = w * w * w;
  w = fminf(fmaxf(w, 0.f), 1.f);
  if (j == nn || r2 > 1.f) w = 0.f;
  float n2 = sqrtf(r2);
  float ninf = fmaxf(fabsf(u0), fmaxf(fabsf(u1), fabsf(u2)));
  float sc = (ninf > 0.f) ? (n2 / (ninf + 1e-12f)) : 0.f;
  float q[3] = {u0 * sc, u1 * sc, u2 * sc};
  int g0[3]; float f[3];
#pragma unroll
  for (int d = 0; d < 3; ++d) {
    float g = fminf(fmaxf((q[d] + 1.f) * 0.5f * 3.f, 0.f), 3.f);
    int gi = (int)floorf(g);
    if (gi > 2) gi = 2;
    g0[d] = gi;
    f[d] = g - (float)gi;
  }
  gw[i] = w; gf0[i] = f[0]; gf1[i] = f[1]; gf2[i] = f[2];
  gbase[i] = (g0[0] * 4 + g0[1]) * 4 + g0[2];
}

// ---------------------------------------------------------------------------
// Binning: one block per point; 64*CIN bins accumulated with ds_add_f32
// ---------------------------------------------------------------------------
template <int CIN>
__global__ void __launch_bounds__(256)
bin_kernel(const float* __restrict__ feat, const int* __restrict__ nbr,
           const float* __restrict__ gw, const float* __restrict__ gf0,
           const float* __restrict__ gf1, const float* __restrict__ gf2,
           const int* __restrict__ gbase, _Float16* __restrict__ binned) {
  __shared__ float bins[64 * CIN];
  int n = blockIdx.x, t = threadIdx.x;
  for (int e = t; e < 64 * CIN; e += 256) bins[e] = 0.f;
  __syncthreads();
  {
    int k = t >> 3, tap = t & 7;          // 32 neighbors * 8 taps = 256 threads
    int i = n * Knbr + k;
    float w = gw[i];
    if (w != 0.f) {
      float f0 = gf0[i], f1 = gf1[i], f2 = gf2[i];
      int dz = (tap >> 2) & 1, dy = (tap >> 1) & 1, dx = tap & 1;
      float wt = w * (dz ? f0 : 1.f - f0) * (dy ? f1 : 1.f - f1) * (dx ? f2 : 1.f - f2);
      int idx = gbase[i] + dz * 16 + dy * 4 + dx;
      int j = nbr[i];
      const float* fj = feat + (size_t)j * CIN;
      float* bp = bins + idx * CIN;
      for (int c = 0; c < CIN; ++c) atomicAdd(&bp[c], wt * fj[c]);
    }
  }
  __syncthreads();
  _Float16* ob = binned + (size_t)n * 64 * CIN;
  for (int e = t; e < 64 * CIN; e += 256) ob[e] = (_Float16)bins[e];
}

// ---------------------------------------------------------------------------
// BatchNorm statistics / apply
// ---------------------------------------------------------------------------
__global__ void zero_f32_kernel(float* p, int n) {
  int i = blockIdx.x * blockDim.x + threadIdx.x;
  if (i < n) p[i] = 0.f;
}

__global__ void __launch_bounds__(256)
bn_stats_kernel(const float* __restrict__ h, float* __restrict__ stats,
                int total, int Cc) {
  __shared__ float ls[128];
  int t = threadIdx.x;
  if (t < 2 * Cc) ls[t] = 0.f;
  __syncthreads();
  for (int e = blockIdx.x * 256 + t; e < total; e += gridDim.x * 256) {
    int c = e % Cc;
    float v = h[e];
    atomicAdd(&ls[c], v);
    atomicAdd(&ls[Cc + c], v * v);
  }
  __syncthreads();
  if (t < 2 * Cc) atomicAdd(&stats[t], ls[t]);
}

__global__ void bn_apply_kernel(float* h, const float* __restrict__ stats,
                                const float* __restrict__ g, const float* __restrict__ be,
                                int total, int Cc, int relu) {
  int e = blockIdx.x * blockDim.x + threadIdx.x;
  if (e >= total) return;
  int c = e % Cc;
  float inv_n = 1.f / (float)Npts;
  float m = stats[c] * inv_n;
  float var = stats[Cc + c] * inv_n - m * m;
  float v = g[c] * (h[e] - m) * rsqrtf(var + 1e-5f) + be[c];
  if (relu) v = fmaxf(v, 0.f);
  h[e] = v;
}

// ---------------------------------------------------------------------------
// Positional-encoding stats (mean/std ddof=1 of pos[:,0]) and h = s + posenc
// ---------------------------------------------------------------------------
__global__ void __launch_bounds__(256)
pestats_kernel(const float* __restrict__ pos, float* __restrict__ out) {
  __shared__ float s1[256], s2[256];
  int t = threadIdx.x;
  float a = 0.f, b = 0.f;
  for (int n = t; n < Npts; n += 256) { float v = pos[n * 3]; a += v; b += v * v; }
  s1[t] = a; s2[t] = b;
  __syncthreads();
  for (int off = 128; off; off >>= 1) {
    if (t < off) { s1[t] += s1[t + off]; s2[t] += s2[t + off]; }
    __syncthreads();
  }
  if (t == 0) {
    float m = s1[0] / (float)Npts;
    float var = (s2[0] - (float)Npts * m * m) / (float)(Npts - 1);
    out[0] = m; out[1] = sqrtf(fmaxf(var, 0.f));
  }
}

__global__ void hpe_kernel(const float* __restrict__ s, const float* __restrict__ pos,
                           const float* __restrict__ pest, _Float16* __restrict__ hf) {
  int e = blockIdx.x * blockDim.x + threadIdx.x;
  if (e >= Npts * Cch) return;
  int n = e >> 5, c = e & 31;
  float xn = (pos[n * 3] - pest[0]) / (pest[1] + 1e-8f);
  int i = c >> 1;
  float div = __expf(-(float)(2 * i) * (9.210340371976184f / 32.f)); // ln(1e4)/C
  float ang = xn * div;
  float pe = (c & 1) ? __cosf(ang) : __sinf(ang);
  hf[e] = (_Float16)(s[e] + pe);
}

// ---------------------------------------------------------------------------
// Flash attention, one wave per (16 query rows, head). HD=8 padded into K=32.
// ---------------------------------------------------------------------------
__global__ void __launch_bounds__(32)
attn_kernel(const _Float16* __restrict__ qf, const _Float16* __restrict__ kf,
            const _Float16* __restrict__ vt, _Float16* __restrict__ of) {
  __shared__ _Float16 pl[16 * 32];
  int row0 = blockIdx.x * 16;
  int h = blockIdx.y;
  int lane = threadIdx.x & 31;
  int laneHi = (lane & 16) ? 1 : 0;

  // Q fragment: full 32-dim rows, zero every dim outside head h.
  v16h q = load_a_frag(qf, Cch, row0, 0);
#pragma unroll
  for (int i = 0; i < 16; ++i) {
    int kd = (i < 8 ? i : i + 8) + (laneHi ? 8 : 0);
    if ((kd >> 3) != h) q[i] = (_Float16)0.f;
  }

  const float scale = 0.35355339059327373f; // 1/sqrt(8)
  float mrun[8], lrun[8];
#pragma unroll
  for (int r = 0; r < 8; ++r) { mrun[r] = -1e30f; lrun[r] = 0.f; }
  v8f acc = {};
  const _Float16* vth = vt + (size_t)h * 16 * Npts;

  for (int key0 = 0; key0 < Npts; key0 += 32) {
    v16h b0 = load_b_frag(kf, Cch, key0, 0);      // keys key0..key0+15
    v16h b1 = load_b_frag(kf, Cch, key0 + 16, 0); // keys key0+16..key0+31
    v8f z = {};
    v8f s0 = wmma_f16(q, b0, z);
    v8f s1 = wmma_f16(q, b1, z);
#pragma unroll
    for (int r = 0; r < 8; ++r) {
      float a0 = s0[r] * scale, a1 = s1[r] * scale;
      float t = fmaxf(a0, a1);
#pragma unroll
      for (int off = 8; off >= 1; off >>= 1) t = fmaxf(t, __shfl_xor(t, off, 32));
      float mnew = fmaxf(mrun[r], t);
      float p0 = __expf(a0 - mnew), p1 = __expf(a1 - mnew);
      float rs = p0 + p1;
#pragma unroll
      for (int off = 8; off >= 1; off >>= 1) rs += __shfl_xor(rs, off, 32);
      float alpha = __expf(mrun[r] - mnew);
      lrun[r] = lrun[r] * alpha + rs;
      acc[r] *= alpha;
      mrun[r] = mnew;
      int row = r + (laneHi ? 8 : 0);
      pl[row * 32 + (lane & 15)]      = (_Float16)p0;
      pl[row * 32 + 16 + (lane & 15)] = (_Float16)p1;
    }
    __syncthreads();
    v16h pa = load_a_frag(pl, 32, 0, 0);          // P: 16 rows x 32 keys
    v16h vb = load_b_frag(vth, Npts, 0, key0);    // V^T: cols = dims (8 real)
    acc = wmma_f16(pa, vb, acc);
    __syncthreads();
  }
  int n = lane & 15;
  if (n < HD) {
#pragma unroll
    for (int r = 0; r < 8; ++r) {
      int row = row0 + r + (laneHi ? 8 : 0);
      of[(size_t)row * Cch + h * HD + n] = (_Float16)(acc[r] / lrun[r]);
    }
  }
}

// ---------------------------------------------------------------------------
// Small elementwise / layout kernels
// ---------------------------------------------------------------------------
__global__ void trbt_kernel(const float* __restrict__ src, _Float16* __restrict__ dst,
                            int Kd, int Nc) {
  int e = blockIdx.x * blockDim.x + threadIdx.x;
  if (e >= Kd * Nc) return;
  int kd = e / Nc, nc = e % Nc;
  dst[(size_t)nc * Kd + kd] = (_Float16)src[e];
}

__global__ void trv_kernel(const _Float16* __restrict__ v, _Float16* __restrict__ vt) {
  int e = blockIdx.x * blockDim.x + threadIdx.x;
  if (e >= Npts * Cch) return;
  int n = e >> 5, d = e & 31;
  vt[(size_t)((d >> 3) * 16 + (d & 7)) * Npts + n] = v[e];
}

__global__ void add2_kernel(const float* a, const float* b, float* c, int n) {
  int e = blockIdx.x * blockDim.x + threadIdx.x;
  if (e < n) c[e] = a[e] + b[e];
}

__global__ void resid_ln_kernel(const float* __restrict__ a, const float* __restrict__ b,
                                const float* __restrict__ g, const float* __restrict__ be,
                                float* __restrict__ outf, _Float16* __restrict__ outh) {
  int n = blockIdx.x * blockDim.x + threadIdx.x;
  if (n >= Npts) return;
  float v[32]; float m = 0.f;
#pragma unroll
  for (int c = 0; c < 32; ++c) { v[c] = a[n * 32 + c] + b[n * 32 + c]; m += v[c]; }
  m *= (1.f / 32.f);
  float var = 0.f;
#pragma unroll
  for (int c = 0; c < 32; ++c) { float d = v[c] - m; var += d * d; }
  var *= (1.f / 32.f);
  float inv = rsqrtf(var + 1e-5f);
#pragma unroll
  for (int c = 0; c < 32; ++c) {
    float o = g[c] * (v[c] - m) * inv + be[c];
    outf[n * 32 + c] = o;
    if (outh) outh[n * 32 + c] = (_Float16)o;
  }
}

__global__ void gate_kernel(const float* __restrict__ x, const float* __restrict__ y,
                            const float* __restrict__ bx, const float* __restrict__ by,
                            float* __restrict__ out, int n) {
  int e = blockIdx.x * blockDim.x + threadIdx.x;
  if (e >= n) return;
  float s = 1.f / (1.f + __expf(-(bx[e] + by[e])));
  out[e] = 2.f * x[e] * s + 2.f * y[e] * (1.f - s);
}

// ---------------------------------------------------------------------------
// Host orchestration
// ---------------------------------------------------------------------------
extern "C" void kernel_launch(void* const* d_in, const int* in_sizes, int n_in,
                              void* d_out, int out_size, void* d_ws, size_t ws_size,
                              hipStream_t stream) {
  (void)in_sizes; (void)n_in; (void)out_size; (void)ws_size;
  const float* x   = (const float*)d_in[0];
  const float* y   = (const float*)d_in[1];
  const float* pos = (const float*)d_in[2];
  const int*   nbr = (const int*)d_in[3];
  auto P = [&](int i) -> const float* { return (const float*)d_in[4 + i]; };
  // params: br*8 + {W1,b1,g1,be1,W2,b2,g2,be2}, then
  // 32:Wq 33:Wk 34:Wv 35:Wo 36:bq 37:bk 38:bv 39:bo
  // 40:ln1_g 41:ln1_b 42:ln2_g 43:ln2_b 44:f_W1 45:f_b1 46:f_W2 47:f_b2

  char* base = (char*)d_ws;
  size_t cur = 0;
  auto alloc = [&](size_t bytes) -> void* {
    void* p = base + cur;
    cur += (bytes + 255) & ~(size_t)255;
    return p;
  };
  const size_t NK = (size_t)Npts * Knbr;
  float*     gw     = (float*)alloc(NK * 4);
  float*     gf0    = (float*)alloc(NK * 4);
  float*     gf1    = (float*)alloc(NK * 4);
  float*     gf2    = (float*)alloc(NK * 4);
  int*       gbase  = (int*)alloc(NK * 4);
  _Float16*  binned = (_Float16*)alloc((size_t)Npts * 4096 * 2);
  _Float16*  btc[8];
  for (int i = 0; i < 8; ++i) btc[i] = (_Float16*)alloc(4096 * 32 * 2);
  _Float16*  bt_q  = (_Float16*)alloc(32 * 32 * 2);
  _Float16*  bt_k  = (_Float16*)alloc(32 * 32 * 2);
  _Float16*  bt_v  = (_Float16*)alloc(32 * 32 * 2);
  _Float16*  bt_o  = (_Float16*)alloc(32 * 32 * 2);
  _Float16*  bt_f1 = (_Float16*)alloc(32 * 128 * 2);
  _Float16*  bt_f2 = (_Float16*)alloc(128 * 32 * 2);
  float*     h64   = (float*)alloc((size_t)Npts * 64 * 4);
  float*     bout[4];
  for (int i = 0; i < 4; ++i) bout[i] = (float*)alloc((size_t)Npts * 32 * 4);
  float*     sbuf  = (float*)alloc((size_t)Npts * 32 * 4);
  float*     stats = (float*)alloc(128 * 4);
  float*     pest  = (float*)alloc(256);
  _Float16*  hf16  = (_Float16*)alloc((size_t)Npts * 32 * 2);
  _Float16*  qf16  = (_Float16*)alloc((size_t)Npts * 32 * 2);
  _Float16*  kf16  = (_Float16*)alloc((size_t)Npts * 32 * 2);
  _Float16*  vf16  = (_Float16*)alloc((size_t)Npts * 32 * 2);
  _Float16*  vt    = (_Float16*)alloc((size_t)HEADS * 16 * Npts * 2);
  _Float16*  of16  = (_Float16*)alloc((size_t)Npts * 32 * 2);
  float*     attnf = (float*)alloc((size_t)Npts * 32 * 4);
  float*     feat1 = (float*)alloc((size_t)Npts * 32 * 4);
  _Float16*  feat1h= (_Float16*)alloc((size_t)Npts * 32 * 2);
  _Float16*  ff1h  = (_Float16*)alloc((size_t)Npts * 128 * 2);
  float*     ff2   = (float*)alloc((size_t)Npts * 32 * 4);
  float*     fused = (float*)alloc((size_t)Npts * 32 * 4);

  auto run_gemm = [&](const _Float16* A, const _Float16* Bt, const float* bias,
                      float* Cf, _Float16* Ch, int Kd, int Nc, int relu) {
    dim3 g(Npts / 16, 1);
    if (Nc == 32)
      gemm16_kernel<2><<<g, 32, 0, stream>>>(A, Bt, bias, Cf, Ch, Kd, relu);
    else if (Nc == 64)
      gemm16_kernel<4><<<g, 32, 0, stream>>>(A, Bt, bias, Cf, Ch, Kd, relu);
    else
      gemm16_kernel<8><<<g, 32, 0, stream>>>(A, Bt, bias, Cf, Ch, Kd, relu);
  };
  auto run_trbt = [&](const float* src, _Float16* dst, int Kd, int Nc) {
    int tot = Kd * Nc;
    trbt_kernel<<<(tot + 255) / 256, 256, 0, stream>>>(src, dst, Kd, Nc);
  };
  auto run_bn = [&](float* h, int Cc, const float* g_, const float* be_, int relu) {
    zero_f32_kernel<<<1, 256, 0, stream>>>(stats, 2 * Cc);
    bn_stats_kernel<<<256, 256, 0, stream>>>(h, stats, Npts * Cc, Cc);
    int total = Npts * Cc;
    bn_apply_kernel<<<(total + 255) / 256, 256, 0, stream>>>(h, stats, g_, be_,
                                                             total, Cc, relu);
  };
  auto run_branch = [&](const float* fin, int br, float* outp) {
    const float* b1  = P(br * 8 + 1);
    const float* g1  = P(br * 8 + 2);
    const float* be1 = P(br * 8 + 3);
    const float* b2  = P(br * 8 + 5);
    const float* g2  = P(br * 8 + 6);
    const float* be2 = P(br * 8 + 7);
    bin_kernel<32><<<Npts, 256, 0, stream>>>(fin, nbr, gw, gf0, gf1, gf2, gbase, binned);
    run_gemm(binned, btc[br * 2 + 0], b1, h64, nullptr, 64 * 32, 64, 0);
    run_bn(h64, 64, g1, be1, 1);
    bin_kernel<64><<<Npts, 256, 0, stream>>>(h64, nbr, gw, gf0, gf1, gf2, gbase, binned);
    run_gemm(binned, btc[br * 2 + 1], b2, outp, nullptr, 64 * 64, 32, 0);
    run_bn(outp, 32, g2, be2, 0);
  };

  // ---- shared geometry + weight layout prep ----
  geom_kernel<<<((int)NK + 255) / 256, 256, 0, stream>>>(pos, nbr, gw, gf0, gf1, gf2, gbase);
  for (int br = 0; br < 4; ++br) {
    run_trbt(P(br * 8 + 0), btc[br * 2 + 0], 2048, 64);
    run_trbt(P(br * 8 + 4), btc[br * 2 + 1], 4096, 32);
  }
  run_trbt(P(32), bt_q, 32, 32);
  run_trbt(P(33), bt_k, 32, 32);
  run_trbt(P(34), bt_v, 32, 32);
  run_trbt(P(35), bt_o, 32, 32);
  run_trbt(P(44), bt_f1, 32, 128);
  run_trbt(P(46), bt_f2, 128, 32);

  // ---- stage 1 branches ----
  run_branch(x, 0, bout[0]);
  run_branch(y, 1, bout[1]);
  add2_kernel<<<(Npts * 32 + 255) / 256, 256, 0, stream>>>(bout[0], bout[1], sbuf, Npts * 32);

  // ---- transformer ----
  pestats_kernel<<<1, 256, 0, stream>>>(pos, pest);
  hpe_kernel<<<(Npts * 32 + 255) / 256, 256, 0, stream>>>(sbuf, pos, pest, hf16);
  run_gemm(hf16, bt_q, P(36), nullptr, qf16, 32, 32, 0);
  run_gemm(hf16, bt_k, P(37), nullptr, kf16, 32, 32, 0);
  run_gemm(hf16, bt_v, P(38), nullptr, vf16, 32, 32, 0);
  zero_f32_kernel<<<(HEADS * 16 * Npts / 2 + 255) / 256, 256, 0, stream>>>(
      (float*)vt, HEADS * 16 * Npts / 2);
  trv_kernel<<<(Npts * 32 + 255) / 256, 256, 0, stream>>>(vf16, vt);
  attn_kernel<<<dim3(Npts / 16, HEADS), 32, 0, stream>>>(qf16, kf16, vt, of16);
  run_gemm(of16, bt_o, P(39), attnf, nullptr, 32, 32, 0);
  resid_ln_kernel<<<(Npts + 255) / 256, 256, 0, stream>>>(sbuf, attnf, P(40), P(41),
                                                          feat1, feat1h);
  run_gemm(feat1h, bt_f1, P(45), nullptr, ff1h, 32, 128, 1);
  run_gemm(ff1h, bt_f2, P(47), ff2, nullptr, 128, 32, 0);
  resid_ln_kernel<<<(Npts + 255) / 256, 256, 0, stream>>>(feat1, ff2, P(42), P(43),
                                                          fused, nullptr);

  // ---- stage 2 branches + gated fusion ----
  run_branch(fused, 2, bout[2]);
  run_branch(fused, 3, bout[3]);
  gate_kernel<<<(Npts * 32 + 255) / 256, 256, 0, stream>>>(x, y, bout[2], bout[3],
                                                           (float*)d_out, Npts * 32);
}
